// GAT_predictor_74543452389833
// MI455X (gfx1250) — compile-verified, matrix operand order checked by reading, not run on hardware
//
#include <hip/hip_runtime.h>
#include <hip/hip_bf16.h>
#include <math.h>

typedef __attribute__((ext_vector_type(16))) _Float16 v16h;
typedef __attribute__((ext_vector_type(8)))  float    v8f;

constexpr int kB = 8, kN = 64, kH = 512, kHeads = 8, kL = 4, kSP = 64, kCLS = 2;
constexpr float kScale = 22.627416997969522f;   // sqrt(512)

// ---------------------------------------------------------------------------
// Generic WMMA GEMM: C[M,Nn] = op(A[M,K] @ W[K,Nn] [+ R])
// flags: bit0 = ReLU, bit1 = add residual R
// 256 threads (8 waves), block tile 128x64, K-stage 64 (8 WMMAs per barrier)
// LDS: 128*64*2 + 64*64*2 = 24 KB (of 320 KB/WGP)
// ---------------------------------------------------------------------------
constexpr int TM = 128, TN = 64, TK = 64;

union Frag {
  v16h  h;
  uint4 q[2];     // two 16-byte contiguous halves of the fragment
};

struct __attribute__((aligned(4))) h2pack { _Float16 lo, hi; };

__global__ __launch_bounds__(256)
void gemm_f16_wmma(const float* __restrict__ A, const float* __restrict__ W,
                   const float* __restrict__ R, float* __restrict__ C,
                   int M, int Nn, int K, int flags)
{
  __shared__ _Float16 As[TM * TK];      // row-major: As[m*TK + k]
  __shared__ _Float16 Bs[TN * TK];      // n-major:  Bs[n*TK + k]
  const int tid  = threadIdx.x;
  const int wv   = tid >> 5;            // wave 0..7 -> 16-row strip
  const int lane = tid & 31;
  const int g    = lane >> 4;           // lane group (K-striping select)
  const int lr   = lane & 15;
  const int m0   = blockIdx.y * TM;
  const int n0   = blockIdx.x * TN;

  v8f acc[4] = {};

  // A staging: thread -> 1 row x 32 cols (8x float4)
  const int arow = tid >> 1;            // 0..127
  const int acb  = (tid & 1) * 32;      // 0 / 32
  // B staging: thread -> 2 k-rows x 8 n-cols (4x float4), stored as k-pairs
  const int bkk  = (tid >> 3) * 2;      // 0..62 (even)
  const int bnb  = (tid & 7) * 8;       // 0..56

  for (int k0 = 0; k0 < K; k0 += TK) {
    // ---- stage A tile (128x64 fp32 -> f16) ----
    const float4* asrc = (const float4*)(A + (size_t)(m0 + arow) * K + k0 + acb);
#pragma unroll
    for (int q4 = 0; q4 < 8; ++q4) {
      float4 f = asrc[q4];
      int c = acb + q4 * 4;
      As[arow*TK + c+0] = (_Float16)f.x;
      As[arow*TK + c+1] = (_Float16)f.y;
      As[arow*TK + c+2] = (_Float16)f.z;
      As[arow*TK + c+3] = (_Float16)f.w;
    }
    // ---- stage B tile (64x64 fp32 -> f16, n-major with packed k-pairs) ----
    {
      const float* w0 = W + (size_t)(k0 + bkk    ) * Nn + n0 + bnb;
      const float* w1 = W + (size_t)(k0 + bkk + 1) * Nn + n0 + bnb;
      const float4 a0 = *(const float4*)(w0);
      const float4 a1 = *(const float4*)(w0 + 4);
      const float4 b0 = *(const float4*)(w1);
      const float4 b1 = *(const float4*)(w1 + 4);
      h2pack* bp = (h2pack*)&Bs[bnb * TK + bkk];
      // n stride in LDS is TK halfs = TK/2 h2packs
      bp[0*(TK/2)] = { (_Float16)a0.x, (_Float16)b0.x };
      bp[1*(TK/2)] = { (_Float16)a0.y, (_Float16)b0.y };
      bp[2*(TK/2)] = { (_Float16)a0.z, (_Float16)b0.z };
      bp[3*(TK/2)] = { (_Float16)a0.w, (_Float16)b0.w };
      bp[4*(TK/2)] = { (_Float16)a1.x, (_Float16)b1.x };
      bp[5*(TK/2)] = { (_Float16)a1.y, (_Float16)b1.y };
      bp[6*(TK/2)] = { (_Float16)a1.z, (_Float16)b1.z };
      bp[7*(TK/2)] = { (_Float16)a1.w, (_Float16)b1.w };
    }
    if (k0 + TK < K)
      __builtin_prefetch(A + (size_t)(m0 + arow) * K + k0 + TK + acb, 0, 1);
    __syncthreads();

    // ---- fragments ----
    // 16-bit 16x32 layout: lane<16 K={0..7,16..23}, lane>=16 K={8..15,24..31};
    // with k-pairs contiguous each half-fragment is 16B contiguous in LDS.
    Frag af[2];
#pragma unroll
    for (int s = 0; s < 2; ++s) {
      const _Float16* ap = &As[(wv * 16 + lr) * TK + s * 32 + g * 8];
      af[s].q[0] = *(const uint4*)(ap);
      af[s].q[1] = *(const uint4*)(ap + 16);
    }
#pragma unroll
    for (int t = 0; t < 4; ++t) {
#pragma unroll
      for (int s = 0; s < 2; ++s) {
        Frag bf;
        const _Float16* bp = &Bs[(t * 16 + lr) * TK + s * 32 + g * 8];
        bf.q[0] = *(const uint4*)(bp);
        bf.q[1] = *(const uint4*)(bp + 16);
        acc[t] = __builtin_amdgcn_wmma_f32_16x16x32_f16(
            false, af[s].h, false, bf.h, (short)0, acc[t], false, false);
      }
    }
    __syncthreads();
  }

  // C/D layout: VGPR r, lanes 0-15 -> M=r, lanes 16-31 -> M=8+r, N=lr
#pragma unroll
  for (int t = 0; t < 4; ++t) {
#pragma unroll
    for (int r = 0; r < 8; ++r) {
      int row = m0 + wv * 16 + g * 8 + r;
      int col = n0 + t * 16 + lr;
      float vv = acc[t][r];
      if (flags & 2) vv += R[(size_t)row * Nn + col];
      if (flags & 1) vv = fmaxf(vv, 0.0f);
      C[(size_t)row * Nn + col] = vv;
    }
  }
}

// ---------------------------------------------------------------------------
// Embedding gathers
// ---------------------------------------------------------------------------
__global__ __launch_bounds__(128)
void atom_embed(const int* __restrict__ ids, const float* __restrict__ emb,
                float* __restrict__ x)
{
  int r = blockIdx.x;                       // B*N rows
  int id = ids[r];
  const float4* src = (const float4*)(emb + (size_t)id * kH);
  float4* dst = (float4*)(x + (size_t)r * kH);
  dst[threadIdx.x] = src[threadIdx.x];      // 128 * float4 = 512 floats
}

__global__ __launch_bounds__(128)
void bond_embed(const int* __restrict__ adj, const float* __restrict__ emb,
                float* __restrict__ bond)
{
  int r = blockIdx.x;                       // B*N*N rows
  int id = adj[r];
  const float4* src = (const float4*)(emb + (size_t)id * kH);
  float4* dst = (float4*)(bond + (size_t)r * kH);
  dst[threadIdx.x] = src[threadIdx.x];
}

// ---------------------------------------------------------------------------
// LayerNorm over last dim (gamma=1, beta=0), one block per row of 512
// ---------------------------------------------------------------------------
__global__ __launch_bounds__(256)
void ln_rows(const float* __restrict__ X, float* __restrict__ Y)
{
  int r = blockIdx.x;
  int tid = threadIdx.x;
  const float* xr = X + (size_t)r * kH;
  float a = xr[tid], b2 = xr[tid + 256];
  __shared__ float s1[256], s2[256];
  s1[tid] = a + b2;
  s2[tid] = a * a + b2 * b2;
  __syncthreads();
  for (int off = 128; off > 0; off >>= 1) {
    if (tid < off) { s1[tid] += s1[tid + off]; s2[tid] += s2[tid + off]; }
    __syncthreads();
  }
  float mean = s1[0] * (1.0f / kH);
  float var  = s2[0] * (1.0f / kH) - mean * mean;
  float inv  = rsqrtf(var + 1e-5f);
  float* yr = Y + (size_t)r * kH;
  yr[tid]       = (a  - mean) * inv;
  yr[tid + 256] = (b2 - mean) * inv;
}

// ---------------------------------------------------------------------------
// Masked attention with bond bias: one block (64 threads) per (b,h,i)
// logits_j = q_i . (k_j + w_ij) / sqrt(H); mask adj==0; o = attn @ v
// ---------------------------------------------------------------------------
__global__ __launch_bounds__(64)
void attn_kernel(const float* __restrict__ q, const float* __restrict__ k,
                 const float* __restrict__ v, const float* __restrict__ ew,
                 const int* __restrict__ adj, float* __restrict__ o)
{
  int bi = blockIdx.x;
  int i = bi % kN;
  int h = (bi / kN) % kHeads;
  int b = bi / (kN * kHeads);
  int j = threadIdx.x;
  __shared__ float attn[kN];
  __shared__ float red[kN];
  const float* qi  = q  + ((size_t)b * kN + i) * kH + h * kSP;
  const float* kj  = k  + ((size_t)b * kN + j) * kH + h * kSP;
  const float* wij = ew + (((size_t)b * kN + i) * kN + j) * kH + h * kSP;
  float s = 0.f;
  for (int d = 0; d < kSP; ++d) s += qi[d] * (kj[d] + wij[d]);
  s /= kScale;
  if (adj[(size_t)b * kN * kN + (size_t)i * kN + j] <= 0) s = -1e30f;
  red[j] = s; __syncthreads();
  for (int off = 32; off > 0; off >>= 1) {
    if (j < off) red[j] = fmaxf(red[j], red[j + off]);
    __syncthreads();
  }
  float mx = red[0]; __syncthreads();
  float e = __expf(s - mx);
  red[j] = e; __syncthreads();
  for (int off = 32; off > 0; off >>= 1) {
    if (j < off) red[j] += red[j + off];
    __syncthreads();
  }
  attn[j] = e / red[0];
  __syncthreads();
  // thread j computes output dim d=j for row i of this head
  float accv = 0.f;
  const float* vb = v + (size_t)b * kN * kH + h * kSP + j;
  for (int jj = 0; jj < kN; ++jj)
    accv += attn[jj] * vb[(size_t)jj * kH];
  o[((size_t)b * kN + i) * kH + h * kSP + j] = accv;
}

// ---------------------------------------------------------------------------
// Edge attention logits -> 3-way softmax weights, one block per (b,i,j)
// ---------------------------------------------------------------------------
__global__ __launch_bounds__(128)
void edge_logits(const float* __restrict__ ew, const float* __restrict__ x,
                 float* __restrict__ wgt)
{
  int r = blockIdx.x;                       // b*N*N + i*N + j
  int j = r % kN;
  int i = (r / kN) % kN;
  int b = r / (kN * kN);
  const float* e  = ew + (size_t)r * kH;
  const float* xi = x + ((size_t)b * kN + i) * kH;
  const float* xj = x + ((size_t)b * kN + j) * kH;
  float ss = 0.f, a0 = 0.f, a1 = 0.f;
  for (int d = threadIdx.x; d < kH; d += 128) {
    float ev = e[d];
    ss += ev * ev;
    a0 += ev * xi[d];
    a1 += ev * xj[d];
  }
  __shared__ float r0[128], r1[128], r2[128];
  r0[threadIdx.x] = ss; r1[threadIdx.x] = a0; r2[threadIdx.x] = a1;
  __syncthreads();
  for (int off = 64; off > 0; off >>= 1) {
    if (threadIdx.x < off) {
      r0[threadIdx.x] += r0[threadIdx.x + off];
      r1[threadIdx.x] += r1[threadIdx.x + off];
      r2[threadIdx.x] += r2[threadIdx.x + off];
    }
    __syncthreads();
  }
  if (threadIdx.x == 0) {
    float l0 = r0[0] / kScale, l1 = r1[0] / kScale, l2 = r2[0] / kScale;
    float mx = fmaxf(l0, fmaxf(l1, l2));
    float e0 = __expf(l0 - mx), e1 = __expf(l1 - mx), e2 = __expf(l2 - mx);
    float inv = 1.0f / (e0 + e1 + e2);
    wgt[(size_t)r * 3 + 0] = e0 * inv;
    wgt[(size_t)r * 3 + 1] = e1 * inv;
    wgt[(size_t)r * 3 + 2] = e2 * inv;
  }
}

// bond = w0*ew + w1*x_i + w2*x_j, float4 elementwise over [B,N,N,H]
__global__ __launch_bounds__(256)
void bond_update(const float* __restrict__ ew, const float* __restrict__ x,
                 const float* __restrict__ wgt, float* __restrict__ bond)
{
  size_t idx = (size_t)blockIdx.x * 256 + threadIdx.x;   // over B*N*N*(H/4)
  size_t r  = idx / (kH / 4);
  int   c4  = (int)(idx % (kH / 4));
  int j = (int)(r % kN);
  int i = (int)((r / kN) % kN);
  int b = (int)(r / ((size_t)kN * kN));
  float w0 = wgt[r * 3 + 0], w1 = wgt[r * 3 + 1], w2 = wgt[r * 3 + 2];
  float4 e  = ((const float4*)(ew + r * kH))[c4];
  float4 xi = ((const float4*)(x + ((size_t)b * kN + i) * kH))[c4];
  float4 xj = ((const float4*)(x + ((size_t)b * kN + j) * kH))[c4];
  float4 ov;
  ov.x = w0 * e.x + w1 * xi.x + w2 * xj.x;
  ov.y = w0 * e.y + w1 * xi.y + w2 * xj.y;
  ov.z = w0 * e.z + w1 * xi.z + w2 * xj.z;
  ov.w = w0 * e.w + w1 * xi.w + w2 * xj.w;
  ((float4*)(bond + r * kH))[c4] = ov;
}

// ---------------------------------------------------------------------------
// Classifier head: cls = tanh(x[:,0,:] @ Wout); out = softmax(cls @ Wpred + b)
// ---------------------------------------------------------------------------
__global__ __launch_bounds__(256)
void head_kernel(const float* __restrict__ x, const float* __restrict__ Wout,
                 const float* __restrict__ Wpred, const float* __restrict__ bpred,
                 float* __restrict__ out)
{
  int b = blockIdx.x;
  const float* xb = x + (size_t)b * kN * kH;      // row 0 of graph b
  __shared__ float cls[kH];
  for (int jj = threadIdx.x; jj < kH; jj += 256) {
    float s = 0.f;
    for (int kk = 0; kk < kH; ++kk) s += xb[kk] * Wout[(size_t)kk * kH + jj];
    cls[jj] = tanhf(s);
  }
  __syncthreads();
  __shared__ float red[256];
  float logit[kCLS];
  for (int c = 0; c < kCLS; ++c) {
    float s = 0.f;
    for (int jj = threadIdx.x; jj < kH; jj += 256) s += cls[jj] * Wpred[(size_t)jj * kCLS + c];
    red[threadIdx.x] = s; __syncthreads();
    for (int off = 128; off > 0; off >>= 1) {
      if (threadIdx.x < off) red[threadIdx.x] += red[threadIdx.x + off];
      __syncthreads();
    }
    logit[c] = red[0] + bpred[c];
    __syncthreads();
  }
  if (threadIdx.x == 0) {
    float mx = fmaxf(logit[0], logit[1]);
    float e0 = __expf(logit[0] - mx), e1 = __expf(logit[1] - mx);
    float inv = 1.0f / (e0 + e1);
    out[(size_t)b * kCLS + 0] = e0 * inv;
    out[(size_t)b * kCLS + 1] = e1 * inv;
  }
}

// ---------------------------------------------------------------------------
extern "C" void kernel_launch(void* const* d_in, const int* in_sizes, int n_in,
                              void* d_out, int out_size, void* d_ws, size_t ws_size,
                              hipStream_t stream)
{
  (void)in_sizes; (void)n_in; (void)out_size; (void)ws_size;
  const int*   input_batch = (const int*)  d_in[0];
  const int*   adj_m       = (const int*)  d_in[1];
  const float* atom_e      = (const float*)d_in[2];
  const float* bond_e      = (const float*)d_in[3];
  const float* Wq    = (const float*)d_in[4];
  const float* Wk    = (const float*)d_in[5];
  const float* Wv    = (const float*)d_in[6];
  const float* Wew   = (const float*)d_in[7];
  const float* Wst   = (const float*)d_in[8];
  const float* Wf1   = (const float*)d_in[9];
  const float* Wf2   = (const float*)d_in[10];
  const float* Woutw = (const float*)d_in[11];
  const float* Wpred = (const float*)d_in[12];
  const float* bpred = (const float*)d_in[13];

  float* p = (float*)d_ws;
  float* bond = p; p += (size_t)kB * kN * kN * kH;   // 67 MB
  float* ew   = p; p += (size_t)kB * kN * kN * kH;   // 67 MB
  float* x    = p; p += (size_t)kB * kN * kH;
  float* xn   = p; p += (size_t)kB * kN * kH;
  float* q    = p; p += (size_t)kB * kN * kH;
  float* kk   = p; p += (size_t)kB * kN * kH;
  float* vv   = p; p += (size_t)kB * kN * kH;
  float* o    = p; p += (size_t)kB * kN * kH;
  float* ar   = p; p += (size_t)kB * kN * kH;
  float* t2   = p; p += (size_t)kB * kN * kH;
  float* f1   = p; p += (size_t)kB * kN * 4 * kH;    // 4 MB
  float* wgt  = p; p += (size_t)kB * kN * kN * 3;

  const int rows = kB * kN;                          // 512

  atom_embed<<<rows, 128, 0, stream>>>(input_batch, atom_e, x);
  bond_embed<<<kB * kN * kN, 128, 0, stream>>>(adj_m, bond_e, bond);

  const dim3 gs(kH / TN, rows / TM);                 // (8, 4)   512x512 GEMMs
  const dim3 ge(kH / TN, (kB * kN * kN) / TM);       // (8, 256) ew GEMM
  const dim3 gf1(4 * kH / TN, rows / TM);            // (32, 4)  FFN1

  for (int l = 0; l < kL; ++l) {
    ln_rows<<<rows, 256, 0, stream>>>(x, xn);
    gemm_f16_wmma<<<gs, 256, 0, stream>>>(xn, Wq + (size_t)l * kH * kH, nullptr, q,  rows, kH, kH, 0);
    gemm_f16_wmma<<<gs, 256, 0, stream>>>(xn, Wk + (size_t)l * kH * kH, nullptr, kk, rows, kH, kH, 0);
    gemm_f16_wmma<<<gs, 256, 0, stream>>>(xn, Wv + (size_t)l * kH * kH, nullptr, vv, rows, kH, kH, 0);
    // dominant GEMM: [B*N*N, H] @ [H, H]
    gemm_f16_wmma<<<ge, 256, 0, stream>>>(bond, Wew + (size_t)l * kH * kH, nullptr, ew,
                                          kB * kN * kN, kH, kH, 0);
    attn_kernel<<<kB * kHeads * kN, 64, 0, stream>>>(q, kk, vv, ew, adj_m, o);
    gemm_f16_wmma<<<gs, 256, 0, stream>>>(o, Wst + (size_t)l * kH * kH, xn, ar, rows, kH, kH, 2);
    ln_rows<<<rows, 256, 0, stream>>>(ar, t2);
    gemm_f16_wmma<<<gf1, 256, 0, stream>>>(t2, Wf1 + (size_t)l * kH * 4 * kH, nullptr, f1,
                                           rows, 4 * kH, kH, 1);
    gemm_f16_wmma<<<gs, 256, 0, stream>>>(f1, Wf2 + (size_t)l * 4 * kH * kH, ar, x,
                                          rows, kH, 4 * kH, 2);
    edge_logits<<<kB * kN * kN, 128, 0, stream>>>(ew, x, wgt);
    bond_update<<<(kB * kN * kN * (kH / 4)) / 256, 256, 0, stream>>>(ew, x, wgt, bond);
  }

  head_kernel<<<kB, 256, 0, stream>>>(x, Woutw, Wpred, bpred, (float*)d_out);
}
